// Rerank_Module_22514218565797
// MI455X (gfx1250) — matrix operand order repaired and checked by reading, not compiled, not run on hardware
//
#include <hip/hip_runtime.h>

// ---------------------------------------------------------------------------
// Problem constants (from the reference)
// ---------------------------------------------------------------------------
#define BQ     8
#define NTOK   64
#define DMODEL 256
#define KSEL   16
#define NHEAD  8
#define NLEV   4
#define NPTS   4
#define DHEAD  32
#define LRR    3
#define HIMG   384
#define WIMG   512
#define HF     96
#define WF     128
#define PFULL  12288          // HF*WF
#define DFF    1024
#define NTK    1024           // NTOK*KSEL
#define MROW   8192           // BQ*NTK
#define SROWS  16320          // P + P/4 + P/16 + P/64

// ---------------------------------------------------------------------------
// CDNA5 vector types
// ---------------------------------------------------------------------------
typedef __attribute__((ext_vector_type(16))) __bf16 v16bf;
typedef __attribute__((ext_vector_type(8)))  __bf16 v8bf;
typedef __attribute__((ext_vector_type(8)))  float  v8f;
typedef __attribute__((ext_vector_type(4)))  unsigned int u32x4;
typedef __attribute__((ext_vector_type(8)))  int    i32x8;
typedef __attribute__((ext_vector_type(4)))  int    i32x4;

__device__ __forceinline__ unsigned short f2bf(float f) {
  unsigned u = __builtin_bit_cast(unsigned, f);
  unsigned r = u + 0x7FFFu + ((u >> 16) & 1u);   // round-to-nearest-even-ish
  return (unsigned short)(r >> 16);
}
__device__ __forceinline__ __bf16 f2bfh(float f) {
  return __builtin_bit_cast(__bf16, f2bf(f));
}

// ---------------------------------------------------------------------------
// Generic batched GEMM:  C[z] = act(A[z] @ B + bias)
// A: (M x Kd) row-major fp32, B: (Kd x N) row-major fp32 (shared), C: (M x N)
//
// CDNA5 pipeline, double-buffered:
//   TDM (tensor_load_to_lds) DMAs the NEXT 128x32 fp32 A tile into LDS while
//   WMMA consumes the CURRENT tile; s_wait_tensorcnt only before the swap
//   barrier. B tiles staged transposed as bf16; fragments assembled with
//   ds_load_b128-wide accesses; math is v_wmma_f32_16x16x32_bf16.
// Block = 128 threads = 4 waves; block tile 128x64; each wave owns 32x64
// (acc[2][4] -> 8 WMMA per K-step).
// ---------------------------------------------------------------------------
#define TILE_M 128
#define TILE_N 64
#define TILE_K 32
#define A_BUF_BYTES (TILE_M * TILE_K * 4)   // 16384

__global__ void __launch_bounds__(128)
gemm_wmma_kernel(const float* __restrict__ A, const float* __restrict__ Bm,
                 const float* __restrict__ bias, float* __restrict__ C,
                 int M, int N, int Kd, long long strideA, long long strideC,
                 int relu)
{
  // sAf assumed at LDS offset 0 (first shared object): TDM descriptor target.
  __shared__ float          sAf[2][TILE_M][TILE_K];   // 2 x 16 KB fp32 A tiles
  __shared__ unsigned short sB[2][TILE_N][TILE_K];    // 2 x 4 KB bf16 B tiles

  const int tid      = threadIdx.x;
  const int lane     = tid & 31;
  const int wid      = tid >> 5;        // wave id 0..3 -> 32-row slice
  const int laneHalf = lane >> 4;       // 0 / 1
  const int lm       = lane & 15;

  const int rowBase = blockIdx.y * TILE_M;
  const int colBase = blockIdx.x * TILE_N;
  const float* Ab = A + (size_t)blockIdx.z * strideA;
  float*       Cb = C + (size_t)blockIdx.z * strideC;

  // ---- helpers -----------------------------------------------------------
  auto issueA = [&](int ktt, int buf) {
#if defined(__gfx1250__)
    // Tensor Data Mover: one DMA descriptor per block, issued by wave 0.
    if (wid == 0) {
      unsigned long long ga =
          (unsigned long long)(size_t)Ab + ((size_t)rowBase * Kd + ktt) * 4ull;
      int remRows = M - rowBase;                      // OOB rows read as zero
      u32x4 g0 = { 1u,                                // count=1 (user D#)
                   (unsigned)(buf * A_BUF_BYTES),     // lds_addr
                   (unsigned)(ga & 0xFFFFFFFFu),
                   (unsigned)((ga >> 32) & 0x1FFFFFFu) | (2u << 30) }; // type=2
      i32x8 g1 = { (int)(2u << 16),                           // data_size = 4 B
                   (int)(((unsigned)Kd & 0xFFFFu) << 16),     // tensor_dim0 lo16
                   (int)(((unsigned)remRows & 0xFFFFu) << 16),// d0 hi | d1 lo16
                   (int)((unsigned)TILE_K << 16),             // d1 hi | tile_dim0
                   (int)TILE_M,                               // tile_dim1 | tile_dim2=0
                   (int)Kd,                                   // dim0_stride lo32
                   0, 0 };
      i32x4 gz = { 0, 0, 0, 0 };
#if __clang_major__ >= 23
      i32x8 gz8 = { 0, 0, 0, 0, 0, 0, 0, 0 };
      __builtin_amdgcn_tensor_load_to_lds(g0, g1, gz, gz, gz8, 0);
#else
      __builtin_amdgcn_tensor_load_to_lds(g0, g1, gz, gz, 0);
#endif
    }
#else
    // host-parse / non-gfx1250 fallback: manual A staging
    for (int i = tid; i < TILE_M * TILE_K; i += 128) {
      int r = i >> 5, c = i & 31;
      int gr = rowBase + r;
      sAf[buf][r][c] = (gr < M) ? Ab[(size_t)gr * Kd + (ktt + c)] : 0.0f;
    }
#endif
  };
  auto waitA = [&]() {
#if defined(__gfx1250__)
    if (wid == 0) __builtin_amdgcn_s_wait_tensorcnt((short)0);
#endif
  };
  auto stageB = [&](int ktt, int buf) {
    // transposed bf16: sB[buf][n][k]; global reads coalesced over n
    for (int i = tid; i < TILE_K * TILE_N; i += 128) {
      int r = i >> 6, c = i & 63;
      int gc = colBase + c;
      float v = (gc < N) ? Bm[(size_t)(ktt + r) * N + gc] : 0.0f;
      sB[buf][c][r] = f2bf(v);
    }
  };

  v8f acc[2][4] = {};

  // ---- prologue: fill buffer 0 ------------------------------------------
  issueA(0, 0);
  stageB(0, 0);
  waitA();
  __syncthreads();

  int cur = 0;
  for (int kt = 0; kt < Kd; kt += TILE_K) {
    const int  nxt  = cur ^ 1;
    const bool more = (kt + TILE_K) < Kd;

    // kick off the NEXT tile's DMA + B staging (overlaps with WMMA below)
    if (more) {
      issueA(kt + TILE_K, nxt);
      stageB(kt + TILE_K, nxt);
      if (kt + 2 * TILE_K < Kd) {
        // global_prefetch_b8 two K-steps ahead (one ahead of the DMA)
        __builtin_prefetch(&Ab[(size_t)(rowBase + tid) * Kd + kt + 2 * TILE_K], 0, 1);
        __builtin_prefetch(&Bm[(size_t)(kt + 2 * TILE_K + (tid & 31)) * N + colBase], 0, 1);
      }
    }

    // ---- compute on the CURRENT tile ------------------------------------
    // A (16x32 bf16): lanes 0-15 -> K 0-7 / 16-23, lanes 16-31 -> K 8-15 / 24-31
    v16bf afrag[2];
    #pragma unroll
    for (int i = 0; i < 2; ++i) {
      int ar = wid * 32 + i * 16 + lm;
      int kb = laneHalf * 8;
      v8f c0 = *(const v8f*)&sAf[cur][ar][kb];        // ds_load_b128 x2
      v8f c1 = *(const v8f*)&sAf[cur][ar][kb + 16];
      #pragma unroll
      for (int e = 0; e < 8; ++e) {
        afrag[i][e]     = f2bfh(c0[e]);
        afrag[i][e + 8] = f2bfh(c1[e]);
      }
    }
    // B (32x16 bf16): lanes 0-15 -> K 0-7 / 16-23, lanes 16-31 -> K 8-15 / 24-31
    v16bf bfrag[4];
    #pragma unroll
    for (int j = 0; j < 4; ++j) {
      int bc = j * 16 + lm;
      int kb = laneHalf * 8;
      const v8bf* bp = (const v8bf*)&sB[cur][bc][kb];
      v8bf lo = bp[0], hi = bp[2];
      bfrag[j] = __builtin_shufflevector(lo, hi,
                    0, 1, 2, 3, 4, 5, 6, 7, 8, 9, 10, 11, 12, 13, 14, 15);
    }

    #pragma unroll
    for (int i = 0; i < 2; ++i)
      #pragma unroll
      for (int j = 0; j < 4; ++j)
        acc[i][j] = __builtin_amdgcn_wmma_f32_16x16x32_bf16(
            false, afrag[i], false, bfrag[j], (short)0, acc[i][j], false, false);

    // swap: DMA + B stores of the next tile must land; current tile already
    // consumed into registers by the WMMAs above.
    if (more) {
      waitA();
      __syncthreads();
    }
    cur = nxt;
  }

  // Epilogue: C layout lane->(M = vgpr + 8*laneHalf, N = lane&15)
  #pragma unroll
  for (int i = 0; i < 2; ++i)
    #pragma unroll
    for (int j = 0; j < 4; ++j) {
      int cc = colBase + j * 16 + lm;
      #pragma unroll
      for (int vv = 0; vv < 8; ++vv) {
        int rr = rowBase + wid * 32 + i * 16 + laneHalf * 8 + vv;
        if (rr < M && cc < N) {
          float x = acc[i][j][vv];
          if (bias) x += bias[cc];
          if (relu) x = fmaxf(x, 0.0f);
          Cb[(size_t)rr * N + cc] = x;
        }
      }
    }
}

// ---------------------------------------------------------------------------
// TDM warm kernel: async tensor load of a 64x32 f32 tile of vW into LDS.
// ---------------------------------------------------------------------------
__global__ void tdm_prefetch_kernel(const float* __restrict__ src, float* __restrict__ sink)
{
  __shared__ float lbuf[2048];   // 8 KB, LDS offset 0
#if defined(__gfx1250__)
  {
    unsigned long long ga = (unsigned long long)(size_t)src;
    u32x4 g0 = { 1u, 0u,
                 (unsigned)(ga & 0xFFFFFFFFu),
                 (unsigned)((ga >> 32) & 0x1FFFFFFu) | (2u << 30) };
    i32x8 g1 = { (int)(2u << 16),
                 (int)(256u << 16),
                 (int)(256u << 16),
                 (int)(32u  << 16),
                 (int)(64u),
                 (int)(256u),
                 0, 0 };
    i32x4 gz = { 0, 0, 0, 0 };
#if __clang_major__ >= 23
    i32x8 gz8 = { 0, 0, 0, 0, 0, 0, 0, 0 };
    __builtin_amdgcn_tensor_load_to_lds(g0, g1, gz, gz, gz8, 0);
#else
    __builtin_amdgcn_tensor_load_to_lds(g0, g1, gz, gz, 0);
#endif
    __builtin_amdgcn_s_wait_tensorcnt((short)0);
  }
#endif
  __syncthreads();
  sink[threadIdx.x] = lbuf[threadIdx.x];
}

// ---------------------------------------------------------------------------
// Top-K (K=16) over P=12288 scores per (b,t); emits p_patch and ref points.
// ---------------------------------------------------------------------------
__global__ void topk_kernel(const float* __restrict__ c_t,
                            float* __restrict__ ppatch, float* __restrict__ ref)
{
  const int row = blockIdx.x;                 // b*NTOK + t
  const int tid = threadIdx.x;
  const float* c = c_t + (size_t)row * PFULL;
  __shared__ float sv[256];
  __shared__ int   si[256];
  __shared__ int   chosen[KSEL];

  for (int k = 0; k < KSEL; ++k) {
    float best = -3.4e38f; int bi = 0x7FFFFFFF;
    for (int i = tid; i < PFULL; i += 256) {
      bool skip = false;
      for (int j = 0; j < k; ++j) if (chosen[j] == i) { skip = true; break; }
      if (skip) continue;
      float v = c[i];
      if (v > best || (v == best && i < bi)) { best = v; bi = i; }
    }
    sv[tid] = best; si[tid] = bi;
    __syncthreads();
    for (int s = 128; s > 0; s >>= 1) {
      if (tid < s) {
        if (sv[tid + s] > sv[tid] ||
            (sv[tid + s] == sv[tid] && si[tid + s] < si[tid])) {
          sv[tid] = sv[tid + s]; si[tid] = si[tid + s];
        }
      }
      __syncthreads();
    }
    if (tid == 0) chosen[k] = si[0];
    __syncthreads();
  }

  for (int k = tid; k < KSEL; k += 256) {
    int id = chosen[k];
    float xi = (float)(id % WF) * 4.0f + 2.0f;
    float yi = (float)(id / WF) * 4.0f + 2.0f;
    size_t o = ((size_t)row * KSEL + k) * 2;
    ppatch[o + 0] = xi;
    ppatch[o + 1] = yi;
    ref[o + 0] = fminf(fmaxf(xi / (float)WIMG, 0.0f), 1.0f);
    ref[o + 1] = fminf(fmaxf(yi / (float)HIMG, 0.0f), 1.0f);
  }
}

// ---------------------------------------------------------------------------
// qk[b, t*K+k, :] = q_t[b, t, :]
// ---------------------------------------------------------------------------
__global__ void bcast_qk_kernel(const float* __restrict__ q_t, float* __restrict__ qk)
{
  int i = blockIdx.x * 256 + threadIdx.x;
  if (i >= MROW * DMODEL) return;
  int d = i & 255;
  int r = i >> 8;           // global qk row
  int bt = r >> 4;          // / KSEL
  qk[i] = q_t[(size_t)bt * DMODEL + d];
}

// ---------------------------------------------------------------------------
// softmax over groups of 16 (per row, per head) in-place
// ---------------------------------------------------------------------------
__global__ void softmax16_kernel(float* __restrict__ aw)
{
  int g = blockIdx.x * 256 + threadIdx.x;
  if (g >= MROW * NHEAD) return;
  float* p = aw + (size_t)(g >> 3) * 128 + (size_t)(g & 7) * 16;
  float mx = -3.4e38f;
  for (int i = 0; i < 16; ++i) mx = fmaxf(mx, p[i]);
  float sm = 0.0f;
  for (int i = 0; i < 16; ++i) { float e = __expf(p[i] - mx); p[i] = e; sm += e; }
  float inv = 1.0f / sm;
  for (int i = 0; i < 16; ++i) p[i] *= inv;
}

// ---------------------------------------------------------------------------
// Multi-scale deformable sampling: attn_in[b,q,c] = sum_{l,p} aw * bilinear(val)
// ---------------------------------------------------------------------------
__global__ void msdeform_kernel(const float* __restrict__ val,
                                const float* __restrict__ offb,
                                const float* __restrict__ awb,
                                const float* __restrict__ ref,
                                float* __restrict__ attn_in)
{
  int i = blockIdx.x * 256 + threadIdx.x;
  if (i >= MROW * DMODEL) return;
  const int c = i & 255;
  const int q = i >> 8;        // global row: b*NTK + tk
  const int b = q >> 10;       // / NTK
  const int h = c >> 5;

  const int hlv[4] = { 96, 48, 24, 12 };
  const int wlv[4] = { 128, 64, 32, 16 };
  const int stv[4] = { 0, 12288, 15360, 16128 };

  const float rx = ref[(size_t)q * 2 + 0];
  const float ry = ref[(size_t)q * 2 + 1];
  const float* off = offb + (size_t)q * 256;
  const float* aw  = awb  + (size_t)q * 128;

  float acc = 0.0f;
  for (int l = 0; l < NLEV; ++l) {
    const int wl = wlv[l], hl = hlv[l], st = stv[l];
    for (int p = 0; p < NPTS; ++p) {
      int m = (h * NLEV + l) * NPTS + p;
      float w  = aw[m];
      float x  = (rx + off[m * 2 + 0] / (float)wl) * (float)wl - 0.5f;
      float y  = (ry + off[m * 2 + 1] / (float)hl) * (float)hl - 0.5f;
      int x0 = (int)floorf(x), y0 = (int)floorf(y);
      float lx = x - (float)x0, ly = y - (float)y0;
      #pragma unroll
      for (int cy = 0; cy < 2; ++cy)
        #pragma unroll
        for (int cx = 0; cx < 2; ++cx) {
          int xi = x0 + cx, yi = y0 + cy;
          float cw = (cx ? lx : 1.0f - lx) * (cy ? ly : 1.0f - ly);
          if (xi >= 0 && xi < wl && yi >= 0 && yi < hl) {
            size_t vrow = (size_t)b * SROWS + st + (size_t)yi * wl + xi;
            acc += w * cw * val[vrow * DMODEL + c];
          }
        }
    }
  }
  attn_in[i] = acc;
}

// ---------------------------------------------------------------------------
// out = LayerNorm(a + b) * g + beta   (one row of 256 per block)
// ---------------------------------------------------------------------------
__global__ void add_ln_kernel(const float* __restrict__ a, const float* __restrict__ b,
                              const float* __restrict__ g, const float* __restrict__ beta,
                              float* __restrict__ out)
{
  const int row = blockIdx.x, tid = threadIdx.x;
  __shared__ float red[256];
  size_t base = (size_t)row * 256;
  float v = a[base + tid] + b[base + tid];
  red[tid] = v; __syncthreads();
  for (int s = 128; s > 0; s >>= 1) { if (tid < s) red[tid] += red[tid + s]; __syncthreads(); }
  float mean = red[0] * (1.0f / 256.0f);
  __syncthreads();
  float d = v - mean;
  red[tid] = d * d; __syncthreads();
  for (int s = 128; s > 0; s >>= 1) { if (tid < s) red[tid] += red[tid + s]; __syncthreads(); }
  float var = red[0] * (1.0f / 256.0f);
  out[base + tid] = d * rsqrtf(var + 1e-5f) * g[tid] + beta[tid];
}

// ---------------------------------------------------------------------------
// concat helpers
// ---------------------------------------------------------------------------
__global__ void concat_qk_q_kernel(const float* __restrict__ qk,
                                   const float* __restrict__ q_t,
                                   float* __restrict__ out)
{
  int i = blockIdx.x * 256 + threadIdx.x;
  if (i >= MROW * 512) return;
  int d = i & 511, r = i >> 9;
  out[i] = (d < 256) ? qk[(size_t)r * 256 + d]
                     : q_t[(size_t)(r >> 4) * 256 + (d - 256)];
}

__global__ void concat_o_q_kernel(const float* __restrict__ o,
                                  const float* __restrict__ q_t,
                                  float* __restrict__ out)
{
  int i = blockIdx.x * 256 + threadIdx.x;
  if (i >= BQ * NTOK * 512) return;
  int d = i & 511, r = i >> 9;
  out[i] = (d < 256) ? o[(size_t)r * 256 + d]
                     : q_t[(size_t)r * 256 + (d - 256)];
}

// ---------------------------------------------------------------------------
// dual logit heads: u = fused@cert_W + cb ; s = fused@sc_W + sb (per row dot)
// ---------------------------------------------------------------------------
__global__ void dual_logit_kernel(const float* __restrict__ x,
                                  const float* __restrict__ cw, const float* __restrict__ cb,
                                  const float* __restrict__ sw, const float* __restrict__ sb,
                                  float* __restrict__ u, float* __restrict__ s)
{
  const int row = blockIdx.x, tid = threadIdx.x;
  __shared__ float rc[256], rs[256];
  float v = x[(size_t)row * 256 + tid];
  rc[tid] = v * cw[tid]; rs[tid] = v * sw[tid];
  __syncthreads();
  for (int st = 128; st > 0; st >>= 1) {
    if (tid < st) { rc[tid] += rc[tid + st]; rs[tid] += rs[tid + st]; }
    __syncthreads();
  }
  if (tid == 0) { u[row] = rc[0] + cb[0]; s[row] = rs[0] + sb[0]; }
}

// ---------------------------------------------------------------------------
// tiny per-token attention: 1 query x K=16 keys, 8 heads of 32
// ---------------------------------------------------------------------------
__global__ void local_attn_kernel(const float* __restrict__ qh,
                                  const float* __restrict__ kh,
                                  const float* __restrict__ vh,
                                  float* __restrict__ o)
{
  const int row = blockIdx.x;     // b*NTOK + t
  const int tid = threadIdx.x;
  __shared__ float sc[NHEAD][KSEL];
  if (tid < NHEAD * KSEL) {
    int h = tid >> 4, k = tid & 15;
    float a = 0.0f;
    for (int d = 0; d < DHEAD; ++d)
      a += qh[(size_t)row * 256 + h * 32 + d] *
           kh[((size_t)row * KSEL + k) * 256 + h * 32 + d];
    sc[h][k] = a * 0.1767766953f;   // 32^-0.5
  }
  __syncthreads();
  if (tid < NHEAD) {
    float mx = -3.4e38f;
    for (int k = 0; k < KSEL; ++k) mx = fmaxf(mx, sc[tid][k]);
    float sm = 0.0f;
    for (int k = 0; k < KSEL; ++k) { float e = __expf(sc[tid][k] - mx); sc[tid][k] = e; sm += e; }
    float inv = 1.0f / sm;
    for (int k = 0; k < KSEL; ++k) sc[tid][k] *= inv;
  }
  __syncthreads();
  int h = tid >> 5, dh = tid & 31;
  float a = 0.0f;
  for (int k = 0; k < KSEL; ++k)
    a += sc[h][k] * vh[((size_t)row * KSEL + k) * 256 + h * 32 + dh];
  o[(size_t)row * 256 + tid] = a;
}

// ---------------------------------------------------------------------------
// Host side
// ---------------------------------------------------------------------------
static inline void launch_gemm(const float* A, const float* B, const float* bias,
                               float* C, int M, int N, int Kd, int batches,
                               long long sA, long long sC, int relu, hipStream_t st)
{
  dim3 grid((N + TILE_N - 1) / TILE_N, (M + TILE_M - 1) / TILE_M, batches);
  gemm_wmma_kernel<<<grid, 128, 0, st>>>(A, B, bias, C, M, N, Kd, sA, sC, relu);
}

extern "C" void kernel_launch(void* const* d_in, const int* in_sizes, int n_in,
                              void* d_out, int out_size, void* d_ws, size_t ws_size,
                              hipStream_t stream)
{
  (void)in_sizes; (void)n_in; (void)out_size; (void)ws_size;

  const float* q_t    = (const float*)d_in[0];
  const float* f4     = (const float*)d_in[1];
  const float* f8     = (const float*)d_in[2];
  const float* f16i   = (const float*)d_in[3];
  const float* f32i   = (const float*)d_in[4];
  const float* c_t    = (const float*)d_in[5];
  const float* off_W  = (const float*)d_in[6];
  const float* off_b  = (const float*)d_in[7];
  const float* attw_W = (const float*)d_in[8];
  const float* attw_b = (const float*)d_in[9];
  const float* vW     = (const float*)d_in[10];
  const float* vb     = (const float*)d_in[11];
  const float* oW     = (const float*)d_in[12];
  const float* ob     = (const float*)d_in[13];
  const float* ln1_g  = (const float*)d_in[14];
  const float* ln1_b  = (const float*)d_in[15];
  const float* ffn1_W = (const float*)d_in[16];
  const float* ffn1_b = (const float*)d_in[17];
  const float* ffn2_W = (const float*)d_in[18];
  const float* ffn2_b = (const float*)d_in[19];
  const float* ln2_g  = (const float*)d_in[20];
  const float* ln2_b  = (const float*)d_in[21];
  const float* fq_W   = (const float*)d_in[22];
  const float* fq_b   = (const float*)d_in[23];
  const float* fk_W   = (const float*)d_in[24];
  const float* fk_b   = (const float*)d_in[25];
  const float* fv_W   = (const float*)d_in[26];
  const float* fv_b   = (const float*)d_in[27];
  const float* fo_W   = (const float*)d_in[28];
  const float* fo_b   = (const float*)d_in[29];
  const float* fln1_g = (const float*)d_in[30];
  const float* fln1_b = (const float*)d_in[31];
  const float* fffn1_W= (const float*)d_in[32];
  const float* fffn1_b= (const float*)d_in[33];
  const float* fffn2_W= (const float*)d_in[34];
  const float* fffn2_b= (const float*)d_in[35];
  const float* fln2_g = (const float*)d_in[36];
  const float* fln2_b = (const float*)d_in[37];
  const float* fus_W  = (const float*)d_in[38];
  const float* fus_b  = (const float*)d_in[39];
  const float* fin_W  = (const float*)d_in[40];
  const float* fin_b  = (const float*)d_in[41];
  const float* cert_W = (const float*)d_in[42];
  const float* cert_b = (const float*)d_in[43];
  const float* sc_W   = (const float*)d_in[44];
  const float* sc_b   = (const float*)d_in[45];

  // output sections (return order: out, p_patch, u_logit, s_logit)
  float* out_main = (float*)d_out;             // 8*64*256
  float* out_pp   = out_main + 131072;         // 8*64*16*2
  float* out_u    = out_pp + 16384;            // 8*64*16
  float* out_s    = out_u + 8192;              // 8*64*16

  // workspace layout (floats)
  float* ws      = (float*)d_ws;
  float* qk      = ws;                         // 2,097,152
  float* refb    = qk + 2097152;               //    16,384
  float* offb    = refb + 16384;               // 2,097,152
  float* awb     = offb + 2097152;             // 1,048,576
  float* attn_in = awb + 1048576;              // 2,097,152
  float* tmp256  = attn_in + 2097152;          // 2,097,152 (attn_out / ffn2 out)
  float* Rbig    = tmp256 + 2097152;           // 33,423,360 region
  float* valb    = Rbig;                       // B x SROWS x D (layer phase)
  float* ff1     = Rbig;                       // reused after msdeform
  // post-loop sub-allocations inside Rbig:
  float* catb  = Rbig;                         // 8192 x 512
  float* fused = Rbig + 4194304;               // 8192 x 256
  float* khb   = fused + 2097152;              // 8192 x 256
  float* vhb   = khb + 2097152;                // 8192 x 256
  float* qhb   = vhb + 2097152;                //  512 x 256
  float* obuf  = qhb + 131072;                 //  512 x 256
  float* o2    = obuf + 131072;                //  512 x 256
  float* fff1  = o2 + 131072;                  //  512 x 1024
  float* fff2  = fff1 + 524288;                //  512 x 256
  float* cat2  = fff2 + 131072;                //  512 x 512
  float* sink  = cat2 + 262144;                //  scratch sink

  // 1) top-K + reference points (+ p_patch output)
  topk_kernel<<<BQ * NTOK, 256, 0, stream>>>(c_t, out_pp, refb);

  // 2) broadcast queries into qk rows
  bcast_qk_kernel<<<(MROW * DMODEL + 255) / 256, 256, 0, stream>>>(q_t, qk);

  // TDM async-tensor warm of the first value-projection weights
  tdm_prefetch_kernel<<<1, 32, 0, stream>>>(vW, sink);

  const float* fseg[4]  = { f4, f8, f16i, f32i };
  const int segRows[4]  = { 12288, 3072, 768, 192 };
  const int segStart[4] = { 0, 12288, 15360, 16128 };

  for (int l = 0; l < LRR; ++l) {
    // offsets & attention-weight logits
    launch_gemm(qk, off_W + (size_t)l * 65536, off_b + l * 256, offb,
                MROW, 256, 256, 1, 0, 0, 0, stream);
    launch_gemm(qk, attw_W + (size_t)l * 32768, attw_b + l * 128, awb,
                MROW, 128, 256, 1, 0, 0, 0, stream);
    softmax16_kernel<<<(MROW * NHEAD + 255) / 256, 256, 0, stream>>>(awb);

    // value projection per pyramid level, batched over B
    for (int s4 = 0; s4 < 4; ++s4) {
      launch_gemm(fseg[s4], vW + (size_t)l * 65536, vb + l * 256,
                  valb + (size_t)segStart[s4] * 256,
                  segRows[s4], 256, 256, BQ,
                  (long long)segRows[s4] * 256, (long long)SROWS * 256, 0, stream);
    }

    // deformable sampling + output projection
    msdeform_kernel<<<(MROW * DMODEL + 255) / 256, 256, 0, stream>>>(
        valb, offb, awb, refb, attn_in);
    launch_gemm(attn_in, oW + (size_t)l * 65536, ob + l * 256, tmp256,
                MROW, 256, 256, 1, 0, 0, 0, stream);
    add_ln_kernel<<<MROW, 256, 0, stream>>>(qk, tmp256,
                                            ln1_g + l * 256, ln1_b + l * 256, qk);

    // FFN
    launch_gemm(qk, ffn1_W + (size_t)l * 262144, ffn1_b + l * 1024, ff1,
                MROW, 1024, 256, 1, 0, 0, 1 /*relu*/, stream);
    launch_gemm(ff1, ffn2_W + (size_t)l * 262144, ffn2_b + l * 256, tmp256,
                MROW, 256, 1024, 1, 0, 0, 0, stream);
    add_ln_kernel<<<MROW, 256, 0, stream>>>(qk, tmp256,
                                            ln2_g + l * 256, ln2_b + l * 256, qk);
  }

  // fusion of refined features with broadcast queries
  concat_qk_q_kernel<<<(MROW * 512 + 255) / 256, 256, 0, stream>>>(qk, q_t, catb);
  launch_gemm(catb, fus_W, fus_b, fused, MROW, 256, 512, 1, 0, 0, 0, stream);

  // certainty / score heads
  dual_logit_kernel<<<MROW, 256, 0, stream>>>(fused, cert_W, cert_b, sc_W, sc_b,
                                              out_u, out_s);

  // small cross-attention over the K candidates
  launch_gemm(q_t,  fq_W, fq_b, qhb, BQ * NTOK, 256, 256, 1, 0, 0, 0, stream);
  launch_gemm(fused, fk_W, fk_b, khb, MROW, 256, 256, 1, 0, 0, 0, stream);
  launch_gemm(fused, fv_W, fv_b, vhb, MROW, 256, 256, 1, 0, 0, 0, stream);
  local_attn_kernel<<<BQ * NTOK, 256, 0, stream>>>(qhb, khb, vhb, obuf);
  launch_gemm(obuf, fo_W, fo_b, o2, BQ * NTOK, 256, 256, 1, 0, 0, 0, stream);
  add_ln_kernel<<<BQ * NTOK, 256, 0, stream>>>(q_t, o2, fln1_g, fln1_b, o2);

  launch_gemm(o2, fffn1_W, fffn1_b, fff1, BQ * NTOK, 1024, 256, 1, 0, 0, 1, stream);
  launch_gemm(fff1, fffn2_W, fffn2_b, fff2, BQ * NTOK, 256, 1024, 1, 0, 0, 0, stream);
  add_ln_kernel<<<BQ * NTOK, 256, 0, stream>>>(o2, fff2, fln2_g, fln2_b, o2);

  // final projection of concat(o, q) -> main output
  concat_o_q_kernel<<<(BQ * NTOK * 512 + 255) / 256, 256, 0, stream>>>(o2, q_t, cat2);
  launch_gemm(cat2, fin_W, fin_b, out_main, BQ * NTOK, 256, 512, 1, 0, 0, 0, stream);
}
